// PlanetoidGAT_27977416966235
// MI455X (gfx1250) — matrix operand order repaired and checked by reading, not compiled, not run on hardware
//
#include <hip/hip_runtime.h>
#include <math.h>

#define NN 50000
#define EE 800000
#define IN_DIM 256
#define F1_DIM 128
#define OUT_DIM 256
#define H1_DIM 64
#define H2_DIM 128
#define NEG_SLOPE 0.01f

typedef __attribute__((ext_vector_type(16))) __bf16 v16bf;
typedef __attribute__((ext_vector_type(8)))  __bf16 v8bf;
typedef __attribute__((ext_vector_type(8)))  float  v8f;

// ---------------- utility kernels ----------------

__global__ void cvt_f32_to_bf16(const float* __restrict__ src, __bf16* __restrict__ dst, long n) {
  long i = (long)blockIdx.x * blockDim.x + threadIdx.x;
  if (i < n) dst[i] = (__bf16)src[i];
}

__global__ void fill_f32(float* __restrict__ p, long n, float v) {
  long i = (long)blockIdx.x * blockDim.x + threadIdx.x;
  if (i < n) p[i] = v;
}

// Convert + transpose + concat one head's weight into Wt[dout_total][din] (bf16),
// and its bias into bcat.
__global__ void prep_wt(const float* __restrict__ W, const float* __restrict__ b,
                        __bf16* __restrict__ Wt, float* __restrict__ bcat,
                        int din, int dh, int h) {
  int t = blockIdx.x * blockDim.x + threadIdx.x;
  if (t < din * dh) {
    int k = t / dh, n = t % dh;
    Wt[(size_t)(h * dh + n) * din + k] = (__bf16)W[(size_t)k * dh + n];
  }
  if (t < dh) bcat[h * dh + t] = b[t];
}

// ---------------- WMMA GEMM: F = X(bf16) @ Wt^T + bias -> bf16 ----------------
// One wave per 16x64 output strip; 4 accumulators share one A fragment.
// k-loop is explicitly double-buffered: loads for step k+32 are issued before
// the WMMAs of step k, so waitcnt never has to fully drain.

__device__ inline v16bf load_frag2(const __bf16* p0, const __bf16* p1) {
  v8bf lo = *(const v8bf*)p0;
  v8bf hi = *(const v8bf*)p1;
  v16bf r;
#pragma unroll
  for (int q = 0; q < 8; ++q) { r[q] = lo[q]; r[q + 8] = hi[q]; }
  return r;
}

__global__ void gemm_bf16_wmma(const __bf16* __restrict__ X, int ldx,
                               const __bf16* __restrict__ Wt,   // [dout_total][Kdim]
                               const float* __restrict__ bias,  // [dout_total]
                               __bf16* __restrict__ Fout, int ldf,
                               int Mtiles, int Ngroups, int Kdim) {
  int w = (int)(((long)blockIdx.x * blockDim.x + threadIdx.x) >> 5);
  if (w >= Mtiles * Ngroups) return;           // wave-uniform: EXEC stays all-ones
  int tm = w / Ngroups, tg = w % Ngroups;      // strip = rows [tm*16,+16) x cols [tg*64,+64)
  int lane = threadIdx.x & 31;
  int l15 = lane & 15, half = lane >> 4;

  // A fragment: row = M, K chunks {8h..8h+7} and {16+8h..16+8h+7}
  const __bf16* arow = X + (size_t)(tm * 16 + l15) * ldx + 8 * half;
  // B fragment: col = N, K chunks {8h..8h+7} and {16+8h..16+8h+7}
  const __bf16* brow[4];
#pragma unroll
  for (int j = 0; j < 4; ++j)
    brow[j] = Wt + (size_t)(tg * 64 + j * 16 + l15) * Kdim + 8 * half;

  v8f acc[4];
#pragma unroll
  for (int j = 0; j < 4; ++j)
    acc[j] = (v8f){0.f, 0.f, 0.f, 0.f, 0.f, 0.f, 0.f, 0.f};

  // prologue: fragments for k = 0
  v16bf A0 = load_frag2(arow, arow + 16);
  v16bf B0[4];
#pragma unroll
  for (int j = 0; j < 4; ++j)
    B0[j] = load_frag2(brow[j], brow[j] + 16);

  int k0 = 0;
  for (; k0 + 32 < Kdim; k0 += 32) {
    int kn = k0 + 32;
    // issue next step's loads first (10x b128 in flight over the WMMAs below)
    v16bf A1 = load_frag2(arow + kn, arow + kn + 16);
    v16bf B1[4];
#pragma unroll
    for (int j = 0; j < 4; ++j)
      B1[j] = load_frag2(brow[j] + kn, brow[j] + kn + 16);
#pragma unroll
    for (int j = 0; j < 4; ++j)
      acc[j] = __builtin_amdgcn_wmma_f32_16x16x32_bf16(
          false, A0, false, B0[j], (short)0, acc[j], false, false);
    A0 = A1;
#pragma unroll
    for (int j = 0; j < 4; ++j) B0[j] = B1[j];
  }
  // epilogue k-step
#pragma unroll
  for (int j = 0; j < 4; ++j)
    acc[j] = __builtin_amdgcn_wmma_f32_16x16x32_bf16(
        false, A0, false, B0[j], (short)0, acc[j], false, false);

#pragma unroll
  for (int j = 0; j < 4; ++j) {
    int ncol = tg * 64 + j * 16 + l15;
    float bn = bias[ncol];
#pragma unroll
    for (int r = 0; r < 8; ++r) {
      int mrow = tm * 16 + r + 8 * half;       // C/D layout: VGPR r -> M = r + 8*half
      Fout[(size_t)mrow * ldf + ncol] = (__bf16)(acc[j][r] + bn);
    }
  }
}

// ---------------- attention scalars per node/head ----------------
__global__ void node_att(const __bf16* __restrict__ Fbf, int ldf, int dh,
                         const float* a1w0, const float* a1b0, const float* a2w0, const float* a2b0,
                         const float* a1w1, const float* a1b1, const float* a2w1, const float* a2b1,
                         float* __restrict__ a1, float* __restrict__ a2) {
  int t = blockIdx.x * blockDim.x + threadIdx.x;
  if (t >= 2 * NN) return;
  int h = t >= NN;
  int i = t - h * NN;
  const float* w1 = h ? a1w1 : a1w0;
  const float* w2 = h ? a2w1 : a2w0;
  float s1 = (h ? a1b1 : a1b0)[0];
  float s2 = (h ? a2b1 : a2b0)[0];
  const __bf16* f = Fbf + (size_t)i * ldf + h * dh;
  for (int d = 0; d < dh; ++d) {
    float fv = (float)f[d];
    s1 += fv * w1[d];
    s2 += fv * w2[d];
  }
  a1[t] = s1;   // layout a[h*NN + i]
  a2[t] = s2;
}

// ---------------- edge sweeps ----------------
__device__ inline void atomicMaxF(float* addr, float v) {
  if (v >= 0.f) atomicMax((int*)addr, __float_as_int(v));
  else          atomicMin((unsigned int*)addr, __float_as_uint(v));
}

__device__ inline float edge_logit(int r, int c, int h,
                                   const float* a1, const float* a2) {
  float v = a1[h * NN + r] + a2[h * NN + c];
  return v > 0.f ? v : NEG_SLOPE * v;
}

__global__ void edge_max(const int* __restrict__ rows, const int* __restrict__ cols,
                         const float* __restrict__ a1, const float* __restrict__ a2,
                         float* __restrict__ m) {
  long t = (long)blockIdx.x * blockDim.x + threadIdx.x;
  if (t >= 2L * EE) return;
  long e = t >> 1; int h = (int)(t & 1);
  int r = rows[e], c = cols[e];
  atomicMaxF(&m[h * NN + r], edge_logit(r, c, h, a1, a2));
}

__global__ void edge_sum(const int* __restrict__ rows, const int* __restrict__ cols,
                         const float* __restrict__ a1, const float* __restrict__ a2,
                         const float* __restrict__ m, float* __restrict__ s) {
  long t = (long)blockIdx.x * blockDim.x + threadIdx.x;
  if (t >= 2L * EE) return;
  long e = t >> 1; int h = (int)(t & 1);
  int r = rows[e], c = cols[e];
  float v = edge_logit(r, c, h, a1, a2);
  atomicAdd(&s[h * NN + r], __expf(v - m[h * NN + r]));
}

// one wave32 per (edge, head); lanes stride the head feature dim
__global__ void edge_agg(const int* __restrict__ rows, const int* __restrict__ cols,
                         const float* __restrict__ a1, const float* __restrict__ a2,
                         const float* __restrict__ m, const float* __restrict__ s,
                         const __bf16* __restrict__ Fbf, int dh,
                         float* __restrict__ agg) {
  int ldf = 2 * dh;
  long gw = ((long)blockIdx.x * blockDim.x + threadIdx.x) >> 5;
  long e = gw >> 1;
  int h = (int)(gw & 1);
  if (e >= EE) return;
  int lane = threadIdx.x & 31;
  int r = rows[e], c = cols[e];
  float v = edge_logit(r, c, h, a1, a2);
  float att = __expf(v - m[h * NN + r]) / s[h * NN + r];
  const __bf16* f = Fbf + (size_t)c * ldf + h * dh;
  float* o = agg + (size_t)r * ldf + h * dh;
  for (int d = lane; d < dh; d += 32)
    atomicAdd(&o[d], att * (float)f[d]);
}

// ---------------- activations ----------------
__global__ void relu_to_bf16(const float* __restrict__ src, __bf16* __restrict__ dst, long n) {
  long i = (long)blockIdx.x * blockDim.x + threadIdx.x;
  if (i < n) dst[i] = (__bf16)fmaxf(src[i], 0.f);
}
__global__ void relu_to_f32(const float* __restrict__ src, float* __restrict__ dst, long n) {
  long i = (long)blockIdx.x * blockDim.x + threadIdx.x;
  if (i < n) dst[i] = fmaxf(src[i], 0.f);
}

// ---------------- host launch ----------------
extern "C" void kernel_launch(void* const* d_in, const int* in_sizes, int n_in,
                              void* d_out, int out_size, void* d_ws, size_t ws_size,
                              hipStream_t stream) {
  (void)in_sizes; (void)n_in; (void)out_size; (void)ws_size;
  const float* feat = (const float*)d_in[0];
  const int*   idx  = (const int*)d_in[1];
  const int* rows = idx;
  const int* cols = idx + EE;

  // params: [gat1_h0, gat1_h1, gat2_h0, gat2_h1] x {W, b, a1_w, a1_b, a2_w, a2_b}
  const float *W[4], *B[4], *A1W[4], *A1B[4], *A2W[4], *A2B[4];
  for (int l = 0; l < 4; ++l) {
    int base = 2 + l * 6;
    W[l]   = (const float*)d_in[base + 0];
    B[l]   = (const float*)d_in[base + 1];
    A1W[l] = (const float*)d_in[base + 2];
    A1B[l] = (const float*)d_in[base + 3];
    A2W[l] = (const float*)d_in[base + 4];
    A2B[l] = (const float*)d_in[base + 5];
  }

  char* wsb = (char*)d_ws;
  size_t off = 0;
  auto take = [&](size_t bytes) -> void* {
    off = (off + 255) & ~(size_t)255;
    void* p = wsb + off; off += bytes; return p;
  };
  __bf16* xbf  = (__bf16*)take((size_t)NN * IN_DIM  * 2);
  __bf16* f1bf = (__bf16*)take((size_t)NN * F1_DIM  * 2);
  __bf16* x1bf = (__bf16*)take((size_t)NN * F1_DIM  * 2);
  __bf16* f2bf = (__bf16*)take((size_t)NN * OUT_DIM * 2);
  __bf16* Wt1  = (__bf16*)take((size_t)F1_DIM  * IN_DIM * 2);
  __bf16* Wt2  = (__bf16*)take((size_t)OUT_DIM * F1_DIM * 2);
  float* b1cat = (float*)take(F1_DIM * 4);
  float* b2cat = (float*)take(OUT_DIM * 4);
  float* agg1  = (float*)take((size_t)NN * F1_DIM  * 4);
  float* agg2  = (float*)take((size_t)NN * OUT_DIM * 4);
  float* a1_1  = (float*)take(2L * NN * 4);
  float* a2_1  = (float*)take(2L * NN * 4);
  float* m_1   = (float*)take(2L * NN * 4);
  float* s_1   = (float*)take(2L * NN * 4);
  float* a1_2  = (float*)take(2L * NN * 4);
  float* a2_2  = (float*)take(2L * NN * 4);
  float* m_2   = (float*)take(2L * NN * 4);
  float* s_2   = (float*)take(2L * NN * 4);

  const int T = 256;
  auto cdiv = [](long a, long b) { return (int)((a + b - 1) / b); };

  // prep: bf16 features, transposed/concat bf16 weights
  cvt_f32_to_bf16<<<cdiv((long)NN * IN_DIM, T), T, 0, stream>>>(feat, xbf, (long)NN * IN_DIM);
  prep_wt<<<cdiv(IN_DIM * H1_DIM, T), T, 0, stream>>>(W[0], B[0], Wt1, b1cat, IN_DIM, H1_DIM, 0);
  prep_wt<<<cdiv(IN_DIM * H1_DIM, T), T, 0, stream>>>(W[1], B[1], Wt1, b1cat, IN_DIM, H1_DIM, 1);
  prep_wt<<<cdiv(F1_DIM * H2_DIM, T), T, 0, stream>>>(W[2], B[2], Wt2, b2cat, F1_DIM, H2_DIM, 0);
  prep_wt<<<cdiv(F1_DIM * H2_DIM, T), T, 0, stream>>>(W[3], B[3], Wt2, b2cat, F1_DIM, H2_DIM, 1);

  // ----- layer 1 -----
  {
    int Mt = NN / 16, Ng = (F1_DIM / 16) / 4;   // 16x64 strips
    gemm_bf16_wmma<<<cdiv((long)Mt * Ng * 32, T), T, 0, stream>>>(
        xbf, IN_DIM, Wt1, b1cat, f1bf, F1_DIM, Mt, Ng, IN_DIM);
    node_att<<<cdiv(2L * NN, T), T, 0, stream>>>(
        f1bf, F1_DIM, H1_DIM,
        A1W[0], A1B[0], A2W[0], A2B[0],
        A1W[1], A1B[1], A2W[1], A2B[1], a1_1, a2_1);
    fill_f32<<<cdiv(2L * NN, T), T, 0, stream>>>(m_1, 2L * NN, -INFINITY);
    fill_f32<<<cdiv(2L * NN, T), T, 0, stream>>>(s_1, 2L * NN, 0.f);
    fill_f32<<<cdiv((long)NN * F1_DIM, T), T, 0, stream>>>(agg1, (long)NN * F1_DIM, 0.f);
    edge_max<<<cdiv(2L * EE, T), T, 0, stream>>>(rows, cols, a1_1, a2_1, m_1);
    edge_sum<<<cdiv(2L * EE, T), T, 0, stream>>>(rows, cols, a1_1, a2_1, m_1, s_1);
    edge_agg<<<cdiv(2L * EE * 32, T), T, 0, stream>>>(rows, cols, a1_1, a2_1, m_1, s_1,
                                                      f1bf, H1_DIM, agg1);
    relu_to_bf16<<<cdiv((long)NN * F1_DIM, T), T, 0, stream>>>(agg1, x1bf, (long)NN * F1_DIM);
  }

  // ----- layer 2 -----
  {
    int Mt = NN / 16, Ng = (OUT_DIM / 16) / 4;  // 16x64 strips
    gemm_bf16_wmma<<<cdiv((long)Mt * Ng * 32, T), T, 0, stream>>>(
        x1bf, F1_DIM, Wt2, b2cat, f2bf, OUT_DIM, Mt, Ng, F1_DIM);
    node_att<<<cdiv(2L * NN, T), T, 0, stream>>>(
        f2bf, OUT_DIM, H2_DIM,
        A1W[2], A1B[2], A2W[2], A2B[2],
        A1W[3], A1B[3], A2W[3], A2B[3], a1_2, a2_2);
    fill_f32<<<cdiv(2L * NN, T), T, 0, stream>>>(m_2, 2L * NN, -INFINITY);
    fill_f32<<<cdiv(2L * NN, T), T, 0, stream>>>(s_2, 2L * NN, 0.f);
    fill_f32<<<cdiv((long)NN * OUT_DIM, T), T, 0, stream>>>(agg2, (long)NN * OUT_DIM, 0.f);
    edge_max<<<cdiv(2L * EE, T), T, 0, stream>>>(rows, cols, a1_2, a2_2, m_2);
    edge_sum<<<cdiv(2L * EE, T), T, 0, stream>>>(rows, cols, a1_2, a2_2, m_2, s_2);
    edge_agg<<<cdiv(2L * EE * 32, T), T, 0, stream>>>(rows, cols, a1_2, a2_2, m_2, s_2,
                                                      f2bf, H2_DIM, agg2);
    relu_to_f32<<<cdiv((long)NN * OUT_DIM, T), T, 0, stream>>>(agg2, (float*)d_out,
                                                               (long)NN * OUT_DIM);
  }
}
